// EdgeConv_9405978378388
// MI455X (gfx1250) — matrix operand order repaired and checked by reading, not compiled, not run on hardware
//
#include <hip/hip_runtime.h>

// ---- problem constants (from reference) ----
#define BATCH 4
#define NPTS  8192
#define DIM   64
#define KNN   16
#define OC    64
#define ROWS  (BATCH*NPTS*KNN)   // 524288 edges
#define TILES (ROWS/16)          // 32768 row tiles
#define NBLK  512
#define TPB   (TILES/NBLK)       // 64 tiles per block
#define TPW   (TPB/8)            // 8 tiles per wave
#define CNT   (NPTS*KNN)         // instance-norm population per (b, o)

typedef _Float16 v16h __attribute__((ext_vector_type(16)));
typedef _Float16 v2h  __attribute__((ext_vector_type(2)));
typedef float    v8f  __attribute__((ext_vector_type(8)));

__device__ __forceinline__ unsigned pk(float a, float b) {
    // v_cvt_pk_rtz_f16_f32; builtin returns <2 x __fp16>, bit-cast to u32
    return __builtin_bit_cast(unsigned, __builtin_amdgcn_cvt_pkrtz(a, b));
}
__device__ __forceinline__ float lkr(float v) { return fmaxf(v, 0.01f * v); }

// =====================================================================
// Stage 1: gather edge features [y - xc, xc] (128), GEMM with W1 (64x128),
// store pre-norm f16 tile, accumulate per-(b,o) sum/sumsq partials.
// =====================================================================
__global__ __launch_bounds__(256) void k_stage1(
    const float* __restrict__ x, const int* __restrict__ ind,
    const float* __restrict__ W1, _Float16* __restrict__ tout,
    float* __restrict__ part)
{
    __shared__ unsigned wlds[64 * 65];      // W1 as f16, [o][c/2], 130-half stride
    __shared__ unsigned astg[8 * 16 * 65];  // per-wave 16 rows x 130 halves
    __shared__ float ssum[64], ssq[64];

    const int tid = threadIdx.x, bid = blockIdx.x;
    for (int p = tid; p < 64 * 64; p += 256) {      // stage W1 -> f16 LDS
        int o = p >> 6, cp = p & 63;
        wlds[o * 65 + cp] = pk(W1[o * 128 + cp * 2], W1[o * 128 + cp * 2 + 1]);
    }
    if (tid < 64) { ssum[tid] = 0.f; ssq[tid] = 0.f; }
    __syncthreads();

    const int lane = tid & 31, wv = tid >> 5;
    const int rr = lane & 15, hh = lane >> 4;
    const int rb = wv * 16 * 65 + rr * 65;          // u32 base of this lane's row
    const float4* xf4 = (const float4*)x;

    for (int it = 0; it < TPW; ++it) {
        const int tile = bid * TPB + wv * TPW + it;
        const int r = tile * 16 + rr;
        const int xcrow = r >> 4;                   // r / KNN -> (b*N+n)
        if (hh == 0) {                              // c = 0..63 : y - xc
            const int idx = ind[r];
            #pragma unroll
            for (int j = 0; j < 16; ++j) {
                float4 y4 = xf4[idx * 16 + j];
                float4 c4 = xf4[xcrow * 16 + j];
                astg[rb + j * 2]     = pk(y4.x - c4.x, y4.y - c4.y);
                astg[rb + j * 2 + 1] = pk(y4.z - c4.z, y4.w - c4.w);
            }
        } else {                                    // c = 64..127 : xc
            #pragma unroll
            for (int j = 0; j < 16; ++j) {
                float4 c4 = xf4[xcrow * 16 + j];
                astg[rb + 32 + j * 2]     = pk(c4.x, c4.y);
                astg[rb + 32 + j * 2 + 1] = pk(c4.z, c4.w);
            }
        }
        __syncthreads();

        v8f acc[4] = {};
        #pragma unroll
        for (int ch = 0; ch < 4; ++ch) {            // K chunks of 32
            union { unsigned u[8]; v16h v; } A;
            #pragma unroll
            for (int v = 0; v < 8; ++v)             // A: K=(v>>2)*16+(v&3)*2+hh*8
                A.u[v] = astg[rb + ch * 16 + (v >> 2) * 8 + hh * 4 + (v & 3)];
            #pragma unroll
            for (int ct = 0; ct < 4; ++ct) {        // N tiles of 16
                union { unsigned u[8]; v16h v; } Bf;
                const int ob = (ct * 16 + rr) * 65 + ch * 16 + hh * 4;
                #pragma unroll
                for (int v = 0; v < 8; ++v) Bf.u[v] = wlds[ob + (v >> 2) * 8 + (v & 3)];
                acc[ct] = __builtin_amdgcn_wmma_f32_16x16x32_f16(
                    false, A.v, false, Bf.v, (short)0, acc[ct], false, false);
            }
        }

        #pragma unroll
        for (int ct = 0; ct < 4; ++ct) {            // store pre-norm f16 + stats
            float s = 0.f, q = 0.f;
            const int o = ct * 16 + rr;
            #pragma unroll
            for (int i = 0; i < 8; ++i) {
                float val = acc[ct][i];
                tout[(tile * 16 + hh * 8 + i) * 64 + o] = (_Float16)val;
                s += val; q += val * val;
            }
            atomicAdd(&ssum[o], s);
            atomicAdd(&ssq[o], q);
        }
        __syncthreads();
    }
    __syncthreads();
    if (tid < 128) {                                // deterministic per-block partial
        const int o = tid & 63;
        part[(bid * 2 + (tid >> 6)) * 64 + o] = (tid < 64) ? ssum[o] : ssq[o];
    }
}

// =====================================================================
// Stages 2/3: normalize+leaky previous pre-norm tile, GEMM with W (64x64),
// store pre-norm f16, accumulate stats partials.
// =====================================================================
__global__ __launch_bounds__(256) void k_stageN(
    const _Float16* __restrict__ tin, const float* __restrict__ W,
    const float* __restrict__ stats, _Float16* __restrict__ tout,
    float* __restrict__ part)
{
    __shared__ unsigned wlds[64 * 33];
    __shared__ unsigned astg[8 * 16 * 33];
    __shared__ float ssum[64], ssq[64], mloc[64], rloc[64];

    const int tid = threadIdx.x, bid = blockIdx.x;
    const int b = bid >> 7;                         // 128 blocks per batch
    for (int p = tid; p < 64 * 32; p += 256) {
        int o = p >> 5, cp = p & 31;
        wlds[o * 33 + cp] = pk(W[o * 64 + cp * 2], W[o * 64 + cp * 2 + 1]);
    }
    if (tid < 64) {
        ssum[tid] = 0.f; ssq[tid] = 0.f;
        mloc[tid] = stats[b * 64 + tid];
        rloc[tid] = stats[256 + b * 64 + tid];
    }
    __syncthreads();

    const int lane = tid & 31, wv = tid >> 5;
    const int rr = lane & 15, hh = lane >> 4;
    const int rb = wv * 16 * 33 + rr * 33;
    const unsigned* tin32 = (const unsigned*)tin;

    for (int it = 0; it < TPW; ++it) {
        const int tile = bid * TPB + wv * TPW + it;
        const int r = tile * 16 + rr;
        #pragma unroll
        for (int j = 0; j < 16; ++j) {              // normalize + leaky -> f16 LDS
            unsigned u = tin32[r * 32 + hh * 16 + j];
            v2h hv = __builtin_bit_cast(v2h, u);
            const int o0 = hh * 32 + j * 2;
            float f0 = lkr(((float)hv.x - mloc[o0]) * rloc[o0]);
            float f1 = lkr(((float)hv.y - mloc[o0 + 1]) * rloc[o0 + 1]);
            astg[rb + hh * 16 + j] = pk(f0, f1);
        }
        __syncthreads();

        v8f acc[4] = {};
        #pragma unroll
        for (int ch = 0; ch < 2; ++ch) {
            union { unsigned u[8]; v16h v; } A;
            #pragma unroll
            for (int v = 0; v < 8; ++v)
                A.u[v] = astg[rb + ch * 16 + (v >> 2) * 8 + hh * 4 + (v & 3)];
            #pragma unroll
            for (int ct = 0; ct < 4; ++ct) {
                union { unsigned u[8]; v16h v; } Bf;
                const int ob = (ct * 16 + rr) * 33 + ch * 16 + hh * 4;
                #pragma unroll
                for (int v = 0; v < 8; ++v) Bf.u[v] = wlds[ob + (v >> 2) * 8 + (v & 3)];
                acc[ct] = __builtin_amdgcn_wmma_f32_16x16x32_f16(
                    false, A.v, false, Bf.v, (short)0, acc[ct], false, false);
            }
        }
        #pragma unroll
        for (int ct = 0; ct < 4; ++ct) {
            float s = 0.f, q = 0.f;
            const int o = ct * 16 + rr;
            #pragma unroll
            for (int i = 0; i < 8; ++i) {
                float val = acc[ct][i];
                tout[(tile * 16 + hh * 8 + i) * 64 + o] = (_Float16)val;
                s += val; q += val * val;
            }
            atomicAdd(&ssum[o], s);
            atomicAdd(&ssq[o], q);
        }
        __syncthreads();
    }
    __syncthreads();
    if (tid < 128) {
        const int o = tid & 63;
        part[(bid * 2 + (tid >> 6)) * 64 + o] = (tid < 64) ? ssum[o] : ssq[o];
    }
}

// ===== finalize: reduce 128 per-block partials (fixed order) -> mean, rsqrt =====
__global__ __launch_bounds__(256) void k_finalize(
    const float* __restrict__ part, float* __restrict__ stats)
{
    const int tid = threadIdx.x;            // tid = b*64 + o
    const int b = tid >> 6, o = tid & 63;
    float s = 0.f, q = 0.f;
    for (int blk = b * 128; blk < b * 128 + 128; ++blk) {
        s += part[(blk * 2) * 64 + o];
        q += part[(blk * 2 + 1) * 64 + o];
    }
    const float inv = 1.0f / (float)CNT;
    float mean = s * inv;
    float var = fmaxf(q * inv - mean * mean, 0.0f);
    stats[tid] = mean;
    stats[256 + tid] = rsqrtf(var + 1e-5f);
}

// ===== final: max over k of raw pre-norm (monotonic), then norm+leaky once =====
__global__ __launch_bounds__(256) void k_final(
    const _Float16* __restrict__ t3, const float* __restrict__ stats,
    float* __restrict__ out)
{
    const int lane = threadIdx.x & 31, wv = threadIdx.x >> 5;
    const int bn = blockIdx.x * 8 + wv;     // (b*N + n)
    const int b = bn >> 13;
    const unsigned* t32 = (const unsigned*)t3;
    float m0 = -3.4e38f, m1 = -3.4e38f;
    #pragma unroll
    for (int kk = 0; kk < 16; ++kk) {
        v2h hv = __builtin_bit_cast(v2h, t32[(bn * 16 + kk) * 32 + lane]);
        m0 = fmaxf(m0, (float)hv.x);
        m1 = fmaxf(m1, (float)hv.y);
    }
    const int o0 = lane * 2;
    float r0 = lkr((m0 - stats[b * 64 + o0])     * stats[256 + b * 64 + o0]);
    float r1 = lkr((m1 - stats[b * 64 + o0 + 1]) * stats[256 + b * 64 + o0 + 1]);
    ((float2*)out)[bn * 32 + lane] = make_float2(r0, r1);
}

extern "C" void kernel_launch(void* const* d_in, const int* in_sizes, int n_in,
                              void* d_out, int out_size, void* d_ws, size_t ws_size,
                              hipStream_t stream) {
    const float* x  = (const float*)d_in[0];
    const int*  ind = (const int*)d_in[1];    // per harness ABI: integer -> const int*
    const float* W1 = (const float*)d_in[2];
    const float* W2 = (const float*)d_in[3];
    const float* W3 = (const float*)d_in[4];
    float* out = (float*)d_out;

    char* ws = (char*)d_ws;
    const size_t TSZ = (size_t)ROWS * 64 * sizeof(_Float16);   // 64 MB per stage buf
    _Float16* tA = (_Float16*)ws;
    _Float16* tB = (_Float16*)(ws + TSZ);
    float* P1 = (float*)(ws + 2 * TSZ);       // [512][2][64] partials per stage
    float* S1 = P1 + NBLK * 2 * 64;           // [mean 256][rsq 256]
    float* P2 = S1 + 512;
    float* S2 = P2 + NBLK * 2 * 64;
    float* P3 = S2 + 512;
    float* S3 = P3 + NBLK * 2 * 64;

    k_stage1  <<<NBLK, 256, 0, stream>>>(x, ind, W1, tA, P1);
    k_finalize<<<1,    256, 0, stream>>>(P1, S1);
    k_stageN  <<<NBLK, 256, 0, stream>>>(tA, W2, S1, tB, P2);
    k_finalize<<<1,    256, 0, stream>>>(P2, S2);
    k_stageN  <<<NBLK, 256, 0, stream>>>(tB, W3, S2, tA, P3);
    k_finalize<<<1,    256, 0, stream>>>(P3, S3);
    k_final   <<<(BATCH * NPTS) / 8, 256, 0, stream>>>(tA, S3, out);
}
